// EdgedMultiHeadAttention_50130858279187
// MI455X (gfx1250) — matrix operand order repaired and checked by reading, not compiled, not run on hardware
//
#include <hip/hip_runtime.h>
#include <hip/hip_bf16.h>

#define D_MODEL 128
#define N_NODES 20000
#define N_EDGES 640000
#define HPAD 136   // LDS row stride in bf16 elements (conflict-free for frag reads)

typedef __attribute__((ext_vector_type(16))) __bf16 v16bf;
typedef __attribute__((ext_vector_type(8)))  float  v8f;
typedef float f32x4 __attribute__((ext_vector_type(4)));

union ABFrag { unsigned int u[8]; v16bf v; };

static __device__ __forceinline__ unsigned short f32_to_bf16_rne(float f) {
    unsigned int u = __float_as_uint(f);
    u += 0x7FFFu + ((u >> 16) & 1u);
    return (unsigned short)(u >> 16);
}

static __device__ __forceinline__ unsigned int pack_bf16x2(float lo, float hi) {
    return (unsigned int)f32_to_bf16_rne(lo) | ((unsigned int)f32_to_bf16_rne(hi) << 16);
}

// A-matrix fragment (16x32 bf16) from an LDS tile, CDNA5 ISA 7.12.2 layout:
// lane L holds row M = L&15; half = L>>4;
// VGPR j (j<4):  K = kbase + half*8 + 2*(j&3)      (+1 in hi 16 bits)
// VGPR j (j>=4): K = kbase + 16 + half*8 + 2*(j&3)
static __device__ __forceinline__ v16bf load_a_frag(const unsigned short* lds,
                                                    int kbase, int lane) {
    ABFrag a;
    const int M = lane & 15;
    const int hw = lane >> 4;
    const unsigned short* row = lds + M * HPAD + kbase + hw * 8;
#pragma unroll
    for (int j = 0; j < 8; ++j) {
        int koff = ((j >= 4) ? 16 : 0) + 2 * (j & 3);
        a.u[j] = *(const unsigned int*)(row + koff);
    }
    return a.v;
}

// B-matrix fragment (32x16 bf16) from transposed weights WT[n][k] (bf16):
// lane L holds column N = L&15; half = L>>4; VGPR v: K = kbase + (v>=4?16:0) + half*8 + 2*(v&3)
static __device__ __forceinline__ v16bf load_b_frag(const unsigned short* wt,
                                                    int tileN, int kbase, int lane) {
    ABFrag b;
    const int n = tileN * 16 + (lane & 15);
    const unsigned short* col = wt + n * D_MODEL + kbase + (lane >> 4) * 8;
#pragma unroll
    for (int v = 0; v < 8; ++v)
        b.u[v] = *(const unsigned int*)(col + ((v >= 4) ? 16 : 0) + 2 * (v & 3));
    return b.v;
}

// MT 16-row tiles sharing one B fragment per K-step (B reuse = MT).
template <int MT>
static __device__ __forceinline__ void gemm_tiles(const unsigned short* Alds,
                                                  const unsigned short* wt,
                                                  int tn, int lane, v8f acc[MT]) {
#pragma unroll
    for (int kt = 0; kt < 4; ++kt) {
        v16bf b = load_b_frag(wt, tn, kt * 32, lane);
#pragma unroll
        for (int mt = 0; mt < MT; ++mt) {
            v16bf a = load_a_frag(Alds + mt * 16 * HPAD, kt * 32, lane);
            acc[mt] = __builtin_amdgcn_wmma_f32_16x16x32_bf16(
                false, a, false, b, (short)0, acc[mt], false, false);
        }
    }
}

// ---------------------------------------------------------------------------
// Kernel: convert 6 weight matrices f32 [k][n] -> bf16 transposed [n][k]
// order in ws: Wq, Wk, Wv, We, Woe, Wox
// ---------------------------------------------------------------------------
__global__ __launch_bounds__(256) void convert_weights_kernel(
    const float* __restrict__ Wq, const float* __restrict__ Wk,
    const float* __restrict__ Wv, const float* __restrict__ We,
    const float* __restrict__ Woe, const float* __restrict__ Wox,
    unsigned short* __restrict__ wt) {
    int t = blockIdx.x * 256 + threadIdx.x;   // 6*16384 threads
    int m = t >> 14;
    int idx = t & 16383;
    int kk = idx >> 7, nn = idx & 127;
    const float* srcs[6] = {Wq, Wk, Wv, We, Woe, Wox};
    wt[(size_t)m * 16384 + nn * 128 + kk] = f32_to_bf16_rne(srcs[m][kk * 128 + nn]);
}

// ---------------------------------------------------------------------------
// Kernel: node projections q,k,v = x@W + b   (32 nodes/block, 8 waves, MT=2)
// ---------------------------------------------------------------------------
__global__ __launch_bounds__(256) void node_proj_kernel(
    const float* __restrict__ x,
    const unsigned short* __restrict__ wtq, const unsigned short* __restrict__ wtk,
    const unsigned short* __restrict__ wtv,
    const float* __restrict__ bq, const float* __restrict__ bk,
    const float* __restrict__ bv,
    float* __restrict__ q, float* __restrict__ k, float* __restrict__ v) {
    __shared__ unsigned short Alds[32 * HPAD];
    const int tid = threadIdx.x;
    const int nodeBase = blockIdx.x * 32;
    for (int i = tid; i < 32 * 32; i += 256) {           // 32 rows x 32 f32x4 chunks
        int r = i >> 5, c4 = (i & 31) * 4;
        f32x4 xv = *(const f32x4*)(x + (size_t)(nodeBase + r) * D_MODEL + c4);
        unsigned int* p = (unsigned int*)(Alds + r * HPAD + c4);
        p[0] = pack_bf16x2(xv.x, xv.y);
        p[1] = pack_bf16x2(xv.z, xv.w);
    }
    __syncthreads();
    const int wave = tid >> 5, lane = tid & 31;
    const int tn = wave;
    const int n = tn * 16 + (lane & 15);
    const int Mbase = (lane >> 4) * 8;
    const unsigned short* wts[3] = {wtq, wtk, wtv};
    const float* bs[3] = {bq, bk, bv};
    float* outs[3] = {q, k, v};
#pragma unroll
    for (int mtx = 0; mtx < 3; ++mtx) {
        v8f acc[2] = {};
        gemm_tiles<2>(Alds, wts[mtx], tn, lane, acc);
        float bias = bs[mtx][n];
#pragma unroll
        for (int mt = 0; mt < 2; ++mt)
#pragma unroll
            for (int r = 0; r < 8; ++r) {
                int row = nodeBase + mt * 16 + Mbase + r;
                outs[mtx][(size_t)row * D_MODEL + n] = acc[mt][r] + bias;
            }
    }
}

// ---------------------------------------------------------------------------
// Kernel: z[dst] += exp(k[src]*q[dst]/sqrt(dk))   (one thread = 4 channels)
// ---------------------------------------------------------------------------
__global__ __launch_bounds__(256) void edge_z_kernel(
    const float* __restrict__ q, const float* __restrict__ k,
    const int* __restrict__ src, const int* __restrict__ dst,
    float* __restrict__ z) {
    long long t = (long long)blockIdx.x * 256 + threadIdx.x;  // E*32 threads
    int e = (int)(t >> 5);
    int c = ((int)t & 31) * 4;
    int s = src[e], d = dst[e];
    const f32x4 kv = *(const f32x4*)(k + (size_t)s * D_MODEL + c);
    const f32x4 qv = *(const f32x4*)(q + (size_t)d * D_MODEL + c);
    float* zp = z + (size_t)d * D_MODEL + c;
    const float sc = 0.25f;  // 1/sqrt(16)
    atomicAdd(zp + 0, __expf(kv.x * qv.x * sc));
    atomicAdd(zp + 1, __expf(kv.y * qv.y * sc));
    atomicAdd(zp + 2, __expf(kv.z * qv.z * sc));
    atomicAdd(zp + 3, __expf(kv.w * qv.w * sc));
}

// ---------------------------------------------------------------------------
// Kernel: fused edge pass. 64 edges/block, 8 waves, MT=4.
//   ep = e@We + be     (WMMA, e loaded non-temporal)
//   m  = exp(k[src]*q[dst]/4)   (recomputed fp32, gathers hit L2)
//   me = m*ep  -> LDS (bf16)
//   x_agg[dst] += me*v[src]/z[dst]    (f32 atomics, L2)
//   out_e = me@Woe + boe  (WMMA, stored non-temporal)
// ---------------------------------------------------------------------------
__global__ __launch_bounds__(256) void edge_main_kernel(
    const float* __restrict__ e,
    const unsigned short* __restrict__ wte, const float* __restrict__ be,
    const unsigned short* __restrict__ wtoe, const float* __restrict__ boe,
    const float* __restrict__ q, const float* __restrict__ k,
    const float* __restrict__ v, const float* __restrict__ z,
    const int* __restrict__ src, const int* __restrict__ dst,
    float* __restrict__ xagg, float* __restrict__ out_e) {
    __shared__ unsigned short Elds[64 * HPAD];
    __shared__ unsigned short Melds[64 * HPAD];
    __shared__ int ssrc[64], sdst[64];
    const int tid = threadIdx.x;
    const int ebase = blockIdx.x * 64;
    if (tid < 64) { ssrc[tid] = src[ebase + tid]; sdst[tid] = dst[ebase + tid]; }
    for (int i = tid; i < 64 * 32; i += 256) {            // 64 rows x 32 f32x4 chunks
        int r = i >> 5, c4 = (i & 31) * 4;
        f32x4 ev = __builtin_nontemporal_load(
            (const f32x4*)(e + (size_t)(ebase + r) * D_MODEL + c4));
        unsigned int* p = (unsigned int*)(Elds + r * HPAD + c4);
        p[0] = pack_bf16x2(ev.x, ev.y);
        p[1] = pack_bf16x2(ev.z, ev.w);
    }
    __syncthreads();
    const int wave = tid >> 5, lane = tid & 31;
    const int tn = wave;
    const int n = tn * 16 + (lane & 15);
    const int hw = lane >> 4;
    const float sc = 0.25f;

    // ep = e @ We (4 M-tiles per wave, B fragment reused 4x)
    v8f acc[4] = {};
    gemm_tiles<4>(Elds, wte, tn, lane, acc);
    const float biasE = be[n];
#pragma unroll
    for (int mt = 0; mt < 4; ++mt) {
#pragma unroll
        for (int r = 0; r < 8; ++r) {
            int row = mt * 16 + hw * 8 + r;   // local edge row 0..63
            int s = ssrc[row], dn = sdst[row];
            float ep = acc[mt][r] + biasE;
            float kk = k[(size_t)s * D_MODEL + n];
            float qq = q[(size_t)dn * D_MODEL + n];
            float m = __expf(kk * qq * sc);
            float me = m * ep;
            Melds[row * HPAD + n] = f32_to_bf16_rne(me);
            float vv = v[(size_t)s * D_MODEL + n];
            float zz = z[(size_t)dn * D_MODEL + n];
            atomicAdd(xagg + (size_t)dn * D_MODEL + n, me * vv / zz);
        }
    }
    __syncthreads();

    // out_e = me @ Woe + boe
    v8f acc2[4] = {};
    gemm_tiles<4>(Melds, wtoe, tn, lane, acc2);
    const float biasO = boe[n];
#pragma unroll
    for (int mt = 0; mt < 4; ++mt)
#pragma unroll
        for (int r = 0; r < 8; ++r) {
            int row = ebase + mt * 16 + hw * 8 + r;
            __builtin_nontemporal_store(acc2[mt][r] + biasO,
                                        out_e + (size_t)row * D_MODEL + n);
        }
}

// ---------------------------------------------------------------------------
// Kernel: out_x = x_agg @ Wox + box   (32 nodes/block, MT=2)
// ---------------------------------------------------------------------------
__global__ __launch_bounds__(256) void out_x_kernel(
    const float* __restrict__ xagg,
    const unsigned short* __restrict__ wtox, const float* __restrict__ box,
    float* __restrict__ out_x) {
    __shared__ unsigned short Alds[32 * HPAD];
    const int tid = threadIdx.x;
    const int nodeBase = blockIdx.x * 32;
    for (int i = tid; i < 32 * 32; i += 256) {
        int r = i >> 5, c4 = (i & 31) * 4;
        f32x4 xv = *(const f32x4*)(xagg + (size_t)(nodeBase + r) * D_MODEL + c4);
        unsigned int* p = (unsigned int*)(Alds + r * HPAD + c4);
        p[0] = pack_bf16x2(xv.x, xv.y);
        p[1] = pack_bf16x2(xv.z, xv.w);
    }
    __syncthreads();
    const int wave = tid >> 5, lane = tid & 31;
    const int tn = wave;
    const int n = tn * 16 + (lane & 15);
    const int Mbase = (lane >> 4) * 8;
    v8f acc[2] = {};
    gemm_tiles<2>(Alds, wtox, tn, lane, acc);
    float bias = box[n];
#pragma unroll
    for (int mt = 0; mt < 2; ++mt)
#pragma unroll
        for (int r = 0; r < 8; ++r) {
            int row = nodeBase + mt * 16 + Mbase + r;
            out_x[(size_t)row * D_MODEL + n] = acc[mt][r] + bias;
        }
}

// ---------------------------------------------------------------------------
// Host launcher
// ---------------------------------------------------------------------------
extern "C" void kernel_launch(void* const* d_in, const int* in_sizes, int n_in,
                              void* d_out, int out_size, void* d_ws, size_t ws_size,
                              hipStream_t stream) {
    const float* x   = (const float*)d_in[0];
    const float* e   = (const float*)d_in[1];
    const float* Wq  = (const float*)d_in[2];  const float* bq  = (const float*)d_in[3];
    const float* Wk  = (const float*)d_in[4];  const float* bk  = (const float*)d_in[5];
    const float* Wv  = (const float*)d_in[6];  const float* bv  = (const float*)d_in[7];
    const float* We  = (const float*)d_in[8];  const float* be  = (const float*)d_in[9];
    const float* Wox = (const float*)d_in[10]; const float* box = (const float*)d_in[11];
    const float* Woe = (const float*)d_in[12]; const float* boe = (const float*)d_in[13];
    const int*   src = (const int*)d_in[14];
    const int*   dst = (const int*)d_in[15];

    float* out_x = (float*)d_out;                                // [N][128]
    float* out_e = out_x + (size_t)N_NODES * D_MODEL;            // [E][128]

    // workspace layout
    char* ws = (char*)d_ws;
    unsigned short* wt = (unsigned short*)ws;         // 6 * 128*128 bf16 = 196608 B
    const size_t NODE_F = (size_t)N_NODES * D_MODEL;  // 2,560,000 floats
    float* q    = (float*)(ws + 196608);
    float* k    = q + NODE_F;
    float* v    = k + NODE_F;
    float* z    = v + NODE_F;
    float* xagg = z + NODE_F;

    unsigned short* wtq  = wt;
    unsigned short* wtk  = wt + 16384;
    unsigned short* wtv  = wt + 2 * 16384;
    unsigned short* wte  = wt + 3 * 16384;
    unsigned short* wtoe = wt + 4 * 16384;
    unsigned short* wtox = wt + 5 * 16384;

    // zero z and xagg (adjacent)
    hipMemsetAsync(z, 0, 2 * NODE_F * sizeof(float), stream);

    convert_weights_kernel<<<6 * 16384 / 256, 256, 0, stream>>>(
        Wq, Wk, Wv, We, Woe, Wox, wt);

    node_proj_kernel<<<N_NODES / 32, 256, 0, stream>>>(
        x, wtq, wtk, wtv, bq, bk, bv, q, k, v);

    edge_z_kernel<<<(N_EDGES * 32) / 256, 256, 0, stream>>>(q, k, src, dst, z);

    edge_main_kernel<<<N_EDGES / 64, 256, 0, stream>>>(
        e, wte, be, wtoe, boe, q, k, v, z, src, dst, xagg, out_e);

    out_x_kernel<<<N_NODES / 32, 256, 0, stream>>>(xagg, wtox, box, out_x);
}
